// FactorNet_6451040878622
// MI455X (gfx1250) — matrix operand order repaired and checked by reading, not compiled, not run on hardware
//
#include <hip/hip_runtime.h>
#include <hip/hip_bf16.h>

// ---------------------------------------------------------------------------
// CDNA5 (gfx1250) fused gather->MLP kernels using v_wmma_f32_16x16x32_f16.
// wave32; each 256-thread block = 8 waves processes a 64-row tile.
// Weights are pre-swizzled into per-lane WMMA B-fragment order so each
// fragment is one contiguous 32B vector load per lane.
// ---------------------------------------------------------------------------

typedef __attribute__((ext_vector_type(16))) _Float16 v16h;
typedef __attribute__((ext_vector_type(8)))  _Float16 v8h;
typedef __attribute__((ext_vector_type(2)))  _Float16 v2h;
typedef __attribute__((ext_vector_type(8)))  float    v8f;

union V16H { v16h v; v8h half8[2]; _Float16 h[16]; };
union V8F  { v8f  v; float f[8]; };

constexpr int TR   = 64;   // rows per block tile
constexpr int HS   = 72;   // hidden LDS row stride in halves (multiple of 8)
constexpr int XPAD = 8;    // input LDS row padding in halves

// A-matrix fragment (16x32 f16) from LDS, ISA 7.12.2 layout:
// lane L: M = L&15 ; halves 0..7 -> K = k0 + 8*(L>=16) + h ; halves 8..15 -> +16.
// Both runs are 16B-aligned -> 2x ds_load_b128.
__device__ inline v16h ldsA(const _Float16* base, int stride, int m0, int k0, int lane) {
  V16H a;
  const _Float16* p = base + (m0 + (lane & 15)) * stride + k0 + ((lane >> 4) << 3);
  a.half8[0] = *(const v8h*)p;
  a.half8[1] = *(const v8h*)(p + 16);
  return a.v;
}

// B-matrix fragment (32x16 f16) from pre-swizzled global weights:
// fragment for (k-tile tk, n-tile tn) is 32 lanes x 16 halves contiguous.
__device__ inline v16h gBswz(const _Float16* W, int ntn, int k0, int n0, int lane) {
  const _Float16* p = W + ((((size_t)((k0 >> 5) * ntn + (n0 >> 4))) * 32 + lane) * 16);
  return *(const v16h*)p;
}

// C/D fragment (16x16 f32): lane L: N = L&15 ; vgpr j -> M = j + 8*(L>=16).
// Bias + ReLU + store f16 into hidden LDS tile.
__device__ inline void stH(_Float16* H, int m0, int n0, v8f cv, const float* bias, int lane) {
  V8F c; c.v = cv;
  int n = n0 + (lane & 15);
  float bb = bias[n];
  int mb = m0 + ((lane >> 4) << 3);
#pragma unroll
  for (int j = 0; j < 8; ++j) {
    float v = c.f[j] + bb;
    H[(mb + j) * HS + n] = (_Float16)(v > 0.f ? v : 0.f);
  }
}

// Gathered-input element fetch (fp32 source -> one value).
__device__ inline float fetch_in(int mode, int R, int col, int n_rows,
                                 const float* __restrict__ atom, int ad,
                                 const float* __restrict__ fr, int fd,
                                 const int* __restrict__ idx, int k, int reverse,
                                 int in_dim) {
  float v = 0.f;
  if (R < n_rows && col < in_dim) {
    if (mode == 0) {
      int ka = k * ad;
      if (col < ka) {
        int j = col / ad, c = col - j * ad;
        if (reverse) j = k - 1 - j;
        v = atom[(size_t)idx[R * k + j] * ad + c];
      } else {
        v = fr[(size_t)R * fd + (col - ka)];
      }
    } else if (mode == 1) {
      if (col < fd) v = fr[(size_t)(R / k) * fd + col];
      else          v = atom[(size_t)idx[R] * ad + (col - fd)];
    } else {
      v = fr[(size_t)R * fd + col];
    }
  }
  return v;
}

// mode 0 (A2F):  row = factor;  in = atom[idx[r][perm 0..k-1]] ++ fr[r]  (reverse flag)
// mode 1 (F2A):  row = edge;    in = fr[r/k] ++ atom[idx_flat[r]]; out via atomicAdd
// mode 2 (DENSE):row = atom;    in = fr[r] (msgs)
__global__ void __launch_bounds__(256)
fused_gather_mlp(int mode, int n_rows,
                 const float* __restrict__ atom, int ad,
                 const float* __restrict__ fr, int fd,
                 const int* __restrict__ idx, int k, int reverse,
                 const _Float16* __restrict__ W1, const float* __restrict__ b1,
                 int in_dim, int in_pad,
                 const _Float16* __restrict__ W2, const float* __restrict__ b2,
                 const _Float16* __restrict__ W3, const float* __restrict__ b3,
                 float* __restrict__ out, int out_ld, int col_off, int acc) {
  extern __shared__ _Float16 smem[];
  _Float16* X  = smem;                      // [TR][in_pad+XPAD]
  _Float16* H1 = X + TR * (in_pad + XPAD);  // [TR][HS]
  _Float16* H2 = H1 + TR * HS;              // [TR][HS]

  const int xs    = in_pad + XPAD;
  const int tid   = threadIdx.x;
  const int lane  = tid & 31;
  const int w     = tid >> 5;
  const int rbase = blockIdx.x * TR;

  // ---- stage gathered input tile into LDS (f32 -> f16), 2 cols per thread ----
  {
    const int cpairs = in_pad >> 1;
    for (int e = tid; e < TR * cpairs; e += 256) {
      int row = e / cpairs, cp = e - row * cpairs;
      int col = cp << 1;
      int R = rbase + row;
      float v0 = fetch_in(mode, R, col,     n_rows, atom, ad, fr, fd, idx, k, reverse, in_dim);
      float v1 = fetch_in(mode, R, col + 1, n_rows, atom, ad, fr, fd, idx, k, reverse, in_dim);
      v2h p; p.x = (_Float16)v0; p.y = (_Float16)v1;
      *(v2h*)(&X[row * xs + col]) = p;   // 4B-aligned packed store
    }
  }
  __syncthreads();

  const int m0 = (w & 3) * 16;

  // ---- layer 1: X[64,in_pad] x W1[in_pad,64] -> H1 ----
  {
    int n0 = (w >> 2) * 32;
    v8f c0 = {}, c1 = {};
    for (int k0 = 0; k0 < in_pad; k0 += 32) {
      v16h a   = ldsA(X, xs, m0, k0, lane);
      v16h bb0 = gBswz(W1, 4, k0, n0, lane);
      v16h bb1 = gBswz(W1, 4, k0, n0 + 16, lane);
      c0 = __builtin_amdgcn_wmma_f32_16x16x32_f16(false, a, false, bb0, (short)0, c0, false, false);
      c1 = __builtin_amdgcn_wmma_f32_16x16x32_f16(false, a, false, bb1, (short)0, c1, false, false);
    }
    stH(H1, m0, n0, c0, b1, lane);
    stH(H1, m0, n0 + 16, c1, b1, lane);
  }
  __syncthreads();

  // ---- layer 2: H1[64,64] x W2[64,64] -> H2 ----
  {
    int n0 = (w >> 2) * 32;
    v8f c0 = {}, c1 = {};
#pragma unroll
    for (int k0 = 0; k0 < 64; k0 += 32) {
      v16h a   = ldsA(H1, HS, m0, k0, lane);
      v16h bb0 = gBswz(W2, 4, k0, n0, lane);
      v16h bb1 = gBswz(W2, 4, k0, n0 + 16, lane);
      c0 = __builtin_amdgcn_wmma_f32_16x16x32_f16(false, a, false, bb0, (short)0, c0, false, false);
      c1 = __builtin_amdgcn_wmma_f32_16x16x32_f16(false, a, false, bb1, (short)0, c1, false, false);
    }
    stH(H2, m0, n0, c0, b2, lane);
    stH(H2, m0, n0 + 16, c1, b2, lane);
  }
  __syncthreads();

  // ---- layer 3: H2[64,64] x W3[64,32] -> out (direct / += / atomic scatter) ----
  {
    int n0 = (w >> 2) * 16;
    v8f c = {};
#pragma unroll
    for (int k0 = 0; k0 < 64; k0 += 32) {
      v16h a  = ldsA(H2, HS, m0, k0, lane);
      v16h bb = gBswz(W3, 2, k0, n0, lane);
      c = __builtin_amdgcn_wmma_f32_16x16x32_f16(false, a, false, bb, (short)0, c, false, false);
    }
    V8F cc; cc.v = c;
    int n = n0 + (lane & 15);
    float bb = b3[n];
    int mb = m0 + ((lane >> 4) << 3);
#pragma unroll
    for (int j = 0; j < 8; ++j) {
      int R = rbase + mb + j;
      if (R < n_rows) {
        float v = cc.f[j] + bb;
        if (mode == 1)      atomicAdd(&out[(size_t)idx[R] * out_ld + col_off + n], v);
        else if (acc)       out[(size_t)R * out_ld + n] += v;
        else                out[(size_t)R * out_ld + n] = v;
      }
    }
  }
}

// fp32 weight [in][out] -> f16 zero-padded, swizzled into WMMA B-fragment order:
// element (k,n): tile (tk=k/32, tn=n/16), lane = ((k%32 / 8)&1)*16 + n%16, h = (k%8) + ((k%32)/16)*8.
// dst index = ((tk*ntn + tn)*32 + lane)*16 + h  (32B contiguous per lane).
__global__ void convert_w_kernel(const float* __restrict__ w, _Float16* __restrict__ o,
                                 int in_dim, int in_pad, int outd, int ntn) {
  int e = blockIdx.x * blockDim.x + threadIdx.x;
  if (e >= in_pad * outd) return;
  int kk = e / outd, n = e - kk * outd;
  int tk = kk >> 5, kr = kk & 31, tn = n >> 4, nn = n & 15;
  int lane = (((kr >> 3) & 1) << 4) | nn;
  int h = (kr & 7) | ((kr >> 4) << 3);
  size_t d = (((size_t)(tk * ntn + tn)) * 32 + lane) * 16 + h;
  o[d] = (_Float16)(kk < in_dim ? w[kk * outd + n] : 0.f);
}

__global__ void zero_kernel(float* p, int n) {
  int e = blockIdx.x * blockDim.x + threadIdx.x;
  if (e < n) p[e] = 0.f;
}

// ---------------------------------------------------------------------------
extern "C" void kernel_launch(void* const* d_in, const int* in_sizes, int n_in,
                              void* d_out, int out_size, void* d_ws, size_t ws_size,
                              hipStream_t stream) {
  (void)in_sizes; (void)n_in; (void)out_size; (void)ws_size;

  const float* element      = (const float*)d_in[0];  // [30000,100]
  const float* bond_repr    = (const float*)d_in[1];  // [30000,1]
  const float* angle_repr   = (const float*)d_in[2];  // [60000,1]
  const float* torsion_repr = (const float*)d_in[3];  // [90000,1]
  const int*   bond_idx     = (const int*)d_in[4];    // [30000,2]
  const int*   angle_idx    = (const int*)d_in[5];    // [60000,3]
  const int*   torsion_idx  = (const int*)d_in[6];    // [90000,4]

  const int NA = 30000, NB = 30000, NG = 60000, NT = 90000;

  size_t off = 0;
  auto carve = [&](size_t bytes) -> void* {
    void* p = (void*)((char*)d_ws + off);
    off = (off + bytes + 255) & ~(size_t)255;
    return p;
  };

  // activation ping-pong buffers (fp32)
  float* frB[2]   = {(float*)carve((size_t)NB * 32 * 4), (float*)carve((size_t)NB * 32 * 4)};
  float* frA[2]   = {(float*)carve((size_t)NG * 32 * 4), (float*)carve((size_t)NG * 32 * 4)};
  float* frT[2]   = {(float*)carve((size_t)NT * 32 * 4), (float*)carve((size_t)NT * 32 * 4)};
  float* atomB[2] = {(float*)carve((size_t)NA * 32 * 4), (float*)carve((size_t)NA * 32 * 4)};
  float* msgs     = (float*)carve((size_t)NA * 96 * 4);

  // f16 swizzled weights for all 5 rounds x 7 MLPs x 3 layers
  struct MlpW { const _Float16 *W1, *W2, *W3; const float *b1, *b2, *b3; int in_dim, in_pad; };
  MlpW mlp[5][7];
  int pi = 7;
  auto conv = [&](const float* w, int in_dim, int outd) -> const _Float16* {
    int in_pad = (in_dim + 31) & ~31;
    _Float16* o = (_Float16*)carve((size_t)in_pad * outd * sizeof(_Float16));
    int tot = in_pad * outd;
    convert_w_kernel<<<(tot + 255) / 256, 256, 0, stream>>>(w, o, in_dim, in_pad, outd, outd / 16);
    return o;
  };
  for (int r = 0; r < 5; ++r) {
    int ad = r ? 32 : 100;
    int fd = r ? 32 : 1;
    int ind[7] = {2 * ad + fd, 3 * ad + fd, 4 * ad + fd, 32 + ad, 32 + ad, 32 + ad, 96};
    for (int m = 0; m < 7; ++m) {
      MlpW& M = mlp[r][m];
      M.in_dim = ind[m];
      M.in_pad = (ind[m] + 31) & ~31;
      M.W1 = conv((const float*)d_in[pi + 0], ind[m], 64); M.b1 = (const float*)d_in[pi + 1];
      M.W2 = conv((const float*)d_in[pi + 2], 64, 64);     M.b2 = (const float*)d_in[pi + 3];
      M.W3 = conv((const float*)d_in[pi + 4], 64, 32);     M.b3 = (const float*)d_in[pi + 5];
      pi += 6;
    }
  }

  auto launch = [&](int mode, int n_rows, const float* atom, int ad, const float* fr,
                    int fd, const int* idx, int k, int rev, const MlpW& M,
                    float* out, int out_ld, int coff, int acc) {
    int blocks = (n_rows + TR - 1) / TR;
    size_t sh = (size_t)(TR * (M.in_pad + XPAD) + 2 * TR * HS) * sizeof(_Float16);
    fused_gather_mlp<<<blocks, 256, sh, stream>>>(
        mode, n_rows, atom, ad, fr, fd, idx, k, rev,
        M.W1, M.b1, M.in_dim, M.in_pad, M.W2, M.b2, M.W3, M.b3,
        out, out_ld, coff, acc);
  };

  for (int r = 0; r < 5; ++r) {
    int ad = r ? 32 : 100;
    int fd = r ? 32 : 1;
    const float* atom_in = (r == 0) ? element : atomB[(r - 1) & 1];
    float* atom_out = (r == 4) ? (float*)d_out : atomB[r & 1];
    const float* fB_in = (r == 0) ? bond_repr    : frB[(r - 1) & 1];
    const float* fA_in = (r == 0) ? angle_repr   : frA[(r - 1) & 1];
    const float* fT_in = (r == 0) ? torsion_repr : frT[(r - 1) & 1];
    float* fB_out = frB[r & 1];
    float* fA_out = frA[r & 1];
    float* fT_out = frT[r & 1];

    // ---- AtomToFactor (bond/angle symmetrized: forward write, reverse add) ----
    launch(0, NB, atom_in, ad, fB_in, fd, bond_idx, 2, 0, mlp[r][0], fB_out, 32, 0, 0);
    launch(0, NB, atom_in, ad, fB_in, fd, bond_idx, 2, 1, mlp[r][0], fB_out, 32, 0, 1);
    launch(0, NG, atom_in, ad, fA_in, fd, angle_idx, 3, 0, mlp[r][1], fA_out, 32, 0, 0);
    launch(0, NG, atom_in, ad, fA_in, fd, angle_idx, 3, 1, mlp[r][1], fA_out, 32, 0, 1);
    launch(0, NT, atom_in, ad, fT_in, fd, torsion_idx, 4, 0, mlp[r][2], fT_out, 32, 0, 0);

    // ---- zero message accumulator ----
    zero_kernel<<<(NA * 96 + 255) / 256, 256, 0, stream>>>(msgs, NA * 96);

    // ---- FactorToAtom: per-edge MLP, atomic segment-sum into msgs ----
    launch(1, NB * 2, atom_in, ad, fB_out, 32, bond_idx,    2, 0, mlp[r][3], msgs, 96, 0,  2);
    launch(1, NG * 3, atom_in, ad, fA_out, 32, angle_idx,   3, 0, mlp[r][4], msgs, 96, 32, 2);
    launch(1, NT * 4, atom_in, ad, fT_out, 32, torsion_idx, 4, 0, mlp[r][5], msgs, 96, 64, 2);

    // ---- combine ----
    launch(2, NA, nullptr, 0, msgs, 96, nullptr, 0, 0, mlp[r][6], atom_out, 32, 0, 0);
  }
}
